// FOAA_SA_4054449127861
// MI455X (gfx1250) — compile-verified, matrix operand order checked by reading, not run on hardware
//
#include <hip/hip_runtime.h>

// FOAA_SA (per-head dim = 1) for MI455X / gfx1250, wave32.
//
// Key identity: softmax_j(Q_i + K_j) == softmax_j(K_j)  (shift invariance),
// so attn is independent of i and out[b,i,:] is the same row for all i.
// Work collapses from O(B*H*S^2) to O(B*S*D) + a 512KB broadcast write.
//
// K/V projection runs on v_wmma_f32_16x16x32_f16 with a split-f16
// (hi+lo) decomposition: 3 WMMAs per tile give ~fp32 accuracy.

typedef __attribute__((ext_vector_type(16))) _Float16 v16h;
typedef __attribute__((ext_vector_type(8)))  float    v8f;
typedef __attribute__((ext_vector_type(4)))  float    v4f;

#define SEQ   1024
#define DMODL 32
#define NROWS 4096   // B*S
#define NCOLS 64     // [K heads 0..31 | V heads 32..63]

// ---------------------------------------------------------------------------
// Kernel 1: KVt[col][row] = (x @ [Wk;Wv]^T + [bk;bv]) transposed.
// One wave (32 threads) per 16x16 output tile. Grid (256, 4).
// A-layout: lane L holds x row (mTile*16 + L%16), K = (L/16)*16 .. +15.
// B-layout: lane L holds Wcat row n = nTile*16 + L%16 (== column of W^T),
//           K = (L/16)*16 .. +15.  Both are 16 contiguous floats -> b128 loads.
// C/D-layout: lane L, vgpr v -> (M = v + 8*(L/16), N = L%16).
// ---------------------------------------------------------------------------
__global__ __launch_bounds__(32)
void foaa_proj_kv_wmma(const float* __restrict__ x,
                       const float* __restrict__ Wk, const float* __restrict__ bk,
                       const float* __restrict__ Wv, const float* __restrict__ bv,
                       float* __restrict__ KVt)
{
    const int lane    = threadIdx.x;   // 0..31, full wave (EXEC all ones)
    const int mTile   = blockIdx.x;    // 0..255
    const int nTile   = blockIdx.y;    // 0..3
    const int halfSel = lane >> 4;     // which 16-wide K half this lane owns
    const int lid     = lane & 15;

    // ---- A: one row of x, 16 contiguous floats, split into f16 hi/lo ----
    const int   aRow  = mTile * 16 + lid;
    const float* aSrc = x + aRow * DMODL + halfSel * 16;
    v4f aq[4];
#pragma unroll
    for (int q = 0; q < 4; ++q) aq[q] = ((const v4f*)aSrc)[q];

    v16h a_hi, a_lo;
#pragma unroll
    for (int i = 0; i < 16; ++i) {
        float f = ((const float*)aq)[i];
        _Float16 h = (_Float16)f;
        a_hi[i] = h;
        a_lo[i] = (_Float16)(f - (float)h);
    }

    // ---- B: row n of Wcat = column n of Wcat^T, 16 contiguous floats ----
    const int   n    = nTile * 16 + lid;                        // 0..63
    const float* wRow = (n < 32) ? (Wk + n * DMODL) : (Wv + (n - 32) * DMODL);
    const float* bSrc = wRow + halfSel * 16;
    v4f bq4[4];
#pragma unroll
    for (int q = 0; q < 4; ++q) bq4[q] = ((const v4f*)bSrc)[q];

    v16h b_hi, b_lo;
#pragma unroll
    for (int i = 0; i < 16; ++i) {
        float f = ((const float*)bq4)[i];
        _Float16 h = (_Float16)f;
        b_hi[i] = h;
        b_lo[i] = (_Float16)(f - (float)h);
    }

    // ---- C seeded with per-column bias (N = lane&15 for every acc VGPR) ----
    const float bias = (n < 32) ? bk[n] : bv[n - 32];
    v8f c;
#pragma unroll
    for (int i = 0; i < 8; ++i) c[i] = bias;

    // ---- split-f16 GEMM: hi*hi + hi*lo + lo*hi  (~fp32 accuracy) ----
    c = __builtin_amdgcn_wmma_f32_16x16x32_f16(false, a_hi, false, b_hi,
                                               (short)0, c, false, false);
    c = __builtin_amdgcn_wmma_f32_16x16x32_f16(false, a_hi, false, b_lo,
                                               (short)0, c, false, false);
    c = __builtin_amdgcn_wmma_f32_16x16x32_f16(false, a_lo, false, b_hi,
                                               (short)0, c, false, false);

    // ---- store transposed: KVt[n][mTile*16 + 8*halfSel + v], v=0..7 ----
    float* dst = KVt + n * NROWS + mTile * 16 + 8 * halfSel;
    v4f lo4 = { c[0], c[1], c[2], c[3] };
    v4f hi4 = { c[4], c[5], c[6], c[7] };
    ((v4f*)dst)[0] = lo4;
    ((v4f*)dst)[1] = hi4;
}

// ---------------------------------------------------------------------------
// Kernel 2: per (b,h): cOut[b*32+h] = sum_j softmax(K[b,:,h])_j * V[b,j,h].
// KVt layout makes the j-scan contiguous. 128 blocks x 256 threads.
// ---------------------------------------------------------------------------
__global__ __launch_bounds__(256)
void foaa_softmax_reduce(const float* __restrict__ KVt, float* __restrict__ cOut)
{
    __shared__ float red[256];
    const int bh = blockIdx.x;           // 0..127
    const int b  = bh >> 5, h = bh & 31;
    const float* K = KVt + h        * NROWS + b * SEQ;
    const float* V = KVt + (32 + h) * NROWS + b * SEQ;
    const int t = threadIdx.x;

    float k0 = K[t], k1 = K[t + 256], k2 = K[t + 512], k3 = K[t + 768];

    // max over j
    red[t] = fmaxf(fmaxf(k0, k1), fmaxf(k2, k3));
    __syncthreads();
    for (int s = 128; s > 0; s >>= 1) {
        if (t < s) red[t] = fmaxf(red[t], red[t + s]);
        __syncthreads();
    }
    const float m = red[0];
    __syncthreads();

    float e0 = expf(k0 - m), e1 = expf(k1 - m);
    float e2 = expf(k2 - m), e3 = expf(k3 - m);
    float v0 = V[t], v1 = V[t + 256], v2 = V[t + 512], v3 = V[t + 768];
    float ls = (e0 + e1) + (e2 + e3);
    float lv = (e0 * v0 + e1 * v1) + (e2 * v2 + e3 * v3);

    red[t] = ls;
    __syncthreads();
    for (int s = 128; s > 0; s >>= 1) {
        if (t < s) red[t] += red[t + s];
        __syncthreads();
    }
    const float ssum = red[0];
    __syncthreads();

    red[t] = lv;
    __syncthreads();
    for (int s = 128; s > 0; s >>= 1) {
        if (t < s) red[t] += red[t + s];
        __syncthreads();
    }
    if (t == 0) cOut[bh] = red[0] / ssum;
}

// ---------------------------------------------------------------------------
// Kernel 3: row[b][d] = bo[d] + sum_h c[b][h]*Wo[d][h]; broadcast over all i.
// Grid (4, 32), 256 threads; each thread writes one float4 (b128 store).
// ---------------------------------------------------------------------------
__global__ __launch_bounds__(256)
void foaa_out_broadcast(const float* __restrict__ c, const float* __restrict__ Wo,
                        const float* __restrict__ bo, float* __restrict__ out)
{
    __shared__ float row[DMODL];
    const int b     = blockIdx.x;   // batch
    const int chunk = blockIdx.y;   // 0..31
    const int t     = threadIdx.x;

    if (t < DMODL) {
        float acc = bo[t];
        const float* cb = c + b * 32;
        const float* w  = Wo + t * DMODL;   // out = c @ Wo^T
#pragma unroll
        for (int h2 = 0; h2 < 32; ++h2) acc += cb[h2] * w[h2];
        row[t] = acc;
    }
    __syncthreads();

    // out[b] spans 8192 float4s; this block covers j = chunk*256 + t.
    const int d4 = t & 7;                    // float4 index within a 32-float row
    v4f val = { row[4 * d4 + 0], row[4 * d4 + 1],
                row[4 * d4 + 2], row[4 * d4 + 3] };
    ((v4f*)out)[b * (SEQ * DMODL / 4) + chunk * 256 + t] = val;
}

// ---------------------------------------------------------------------------
// Launch. Inputs: x, Wq, bq, Wk, bk, Wv, bv, Wo, bo  (Wq/bq provably unused:
// softmax shift-invariance removes Q from the result).
// Workspace: KVt 64*4096 floats (1 MB) + c 128 floats.
// ---------------------------------------------------------------------------
extern "C" void kernel_launch(void* const* d_in, const int* in_sizes, int n_in,
                              void* d_out, int out_size, void* d_ws, size_t ws_size,
                              hipStream_t stream) {
    (void)in_sizes; (void)n_in; (void)out_size; (void)ws_size;
    const float* x  = (const float*)d_in[0];
    const float* Wk = (const float*)d_in[3];
    const float* bk = (const float*)d_in[4];
    const float* Wv = (const float*)d_in[5];
    const float* bv = (const float*)d_in[6];
    const float* Wo = (const float*)d_in[7];
    const float* bo = (const float*)d_in[8];

    float* KVt = (float*)d_ws;
    float* c   = (float*)((char*)d_ws + (size_t)NCOLS * NROWS * sizeof(float));
    float* out = (float*)d_out;

    foaa_proj_kv_wmma<<<dim3(256, 4), 32, 0, stream>>>(x, Wk, bk, Wv, bv, KVt);
    foaa_softmax_reduce<<<128, 256, 0, stream>>>(KVt, c);
    foaa_out_broadcast<<<dim3(4, 32), 256, 0, stream>>>(c, Wo, bo, out);
}
